// Qwen3MoeSparseMoeBlock_58600533787453
// MI455X (gfx1250) — matrix operand order repaired and checked by reading, not compiled, not run on hardware
//
#include <hip/hip_runtime.h>
#include <hip/hip_bf16.h>

#define HID   1024
#define INTER 768
#define NTOK  2048
#define NEXP  32
#define TOPK  4

typedef __attribute__((ext_vector_type(16))) __bf16 v16bf;
typedef __attribute__((ext_vector_type(8)))  float  v8f;
typedef unsigned short u16;

union FragAB {
    v16bf v;
    uint4 q[2];
    u16   u[16];
};

__device__ __forceinline__ u16 f2bf(float f) {
    __bf16 h = (__bf16)f;                 // native cvt (RNE) if HW supports it
    return __builtin_bit_cast(u16, h);
}

// ---------------------------------------------------------------------------
// Zero output accumulator + per-expert counters
// ---------------------------------------------------------------------------
__global__ void moe_zero(float* __restrict__ out, int* __restrict__ cnt) {
    int idx = blockIdx.x * 256 + threadIdx.x;
    if (idx < NTOK * HID) out[idx] = 0.f;
    if (idx < NEXP) cnt[idx] = 0;
}

// ---------------------------------------------------------------------------
// Convert x fp32 -> bf16 (row-major, A-side operand)
// ---------------------------------------------------------------------------
__global__ __launch_bounds__(256) void moe_cvt_x(const float* __restrict__ x,
                                                 u16* __restrict__ xbf) {
    int i = (blockIdx.x * 256 + threadIdx.x) * 8;
    float4 a = *(const float4*)&x[i];
    float4 b = *(const float4*)&x[i + 4];
    u16 o[8] = { f2bf(a.x), f2bf(a.y), f2bf(a.z), f2bf(a.w),
                 f2bf(b.x), f2bf(b.y), f2bf(b.z), f2bf(b.w) };
    *(uint4*)&xbf[i] = *(uint4*)o;
}

// ---------------------------------------------------------------------------
// Transpose + convert one expert weight slab: dst[c][r](bf16) = src[r][c](f32)
// grid = (C/64, R/64, NEXP), 256 threads, LDS-tiled so both sides coalesce.
// After this, B fragments are K-contiguous: two 16B loads per lane.
// ---------------------------------------------------------------------------
__global__ __launch_bounds__(256) void moe_transpose_cvt(
    const float* __restrict__ src, u16* __restrict__ dst, int R, int C) {
    __shared__ __align__(16) u16 T[64][72];   // 72*2=144B row stride (16B mult)
    const int e  = blockIdx.z;
    const int c0 = blockIdx.x * 64;
    const int r0 = blockIdx.y * 64;
    const float* s = src + (size_t)e * R * C;
    u16* d = dst + (size_t)e * R * C;
    const int tid  = threadIdx.x;
    const int col4 = (tid & 15) * 4;
    const int row  = tid >> 4;
#pragma unroll
    for (int rr = 0; rr < 64; rr += 16) {
        float4 v = *(const float4*)&s[(size_t)(r0 + row + rr) * C + c0 + col4];
        T[col4 + 0][row + rr] = f2bf(v.x);
        T[col4 + 1][row + rr] = f2bf(v.y);
        T[col4 + 2][row + rr] = f2bf(v.z);
        T[col4 + 3][row + rr] = f2bf(v.w);
    }
    __syncthreads();
    const int c    = tid >> 2;
    const int rseg = (tid & 3) * 16;
    uint4 q0 = *(const uint4*)&T[c][rseg];
    uint4 q1 = *(const uint4*)&T[c][rseg + 8];
    u16* drow = &d[(size_t)(c0 + c) * R + r0 + rseg];
    *(uint4*)drow = q0;
    *(uint4*)(drow + 8) = q1;
}

// ---------------------------------------------------------------------------
// Router: one wave32 per token, lane == expert. topk(softmax) renormalized ==
// topk(exp(logit-max)) renormalized -> skip the softmax denominator.
// ---------------------------------------------------------------------------
__global__ __launch_bounds__(256) void moe_router(
    const float* __restrict__ x, const float* __restrict__ gw,
    int* __restrict__ cnt, int* __restrict__ tokmap, float* __restrict__ wmap) {

    const int lane = threadIdx.x & 31;
    const int t = blockIdx.x * 8 + (threadIdx.x >> 5);

    const float* xr = x + (size_t)t * HID;
    const float* gr = gw + (size_t)lane * HID;
    float acc = 0.f;
#pragma unroll 8
    for (int h = 0; h < HID; ++h) acc = fmaf(xr[h], gr[h], acc);

    float m = acc;
    for (int off = 16; off; off >>= 1) m = fmaxf(m, __shfl_xor(m, off, 32));
    float pv = __expf(acc - m);

    float selW[TOPK]; int selE[TOPK]; float wsum = 0.f;
#pragma unroll
    for (int it = 0; it < TOPK; ++it) {
        float bv = pv; int bi = lane;
        for (int off = 16; off; off >>= 1) {
            float ov = __shfl_xor(bv, off, 32);
            int   oi = __shfl_xor(bi, off, 32);
            if (ov > bv || (ov == bv && oi < bi)) { bv = ov; bi = oi; }
        }
        selW[it] = bv; selE[it] = bi; wsum += bv;
        if (lane == bi) pv = -1.f;
    }

    if (lane == 0) {
        float inv = 1.f / wsum;
#pragma unroll
        for (int it = 0; it < TOPK; ++it) {
            int e = selE[it];
            int slot = atomicAdd(&cnt[e], 1);
            tokmap[e * NTOK + slot] = t;
            wmap[e * NTOK + slot] = selW[it] * inv;
        }
    }
}

// ---------------------------------------------------------------------------
// Exclusive prefix sum of 32 expert counts (trivial, one thread)
// ---------------------------------------------------------------------------
__global__ void moe_offsets(const int* __restrict__ cnt, int* __restrict__ off) {
    if (threadIdx.x == 0) {
        int s = 0;
#pragma unroll
        for (int e = 0; e < NEXP; ++e) { off[e] = s; s += cnt[e]; }
    }
}

// ---------------------------------------------------------------------------
// Gate+Up GEMM: block = (expert, 64-token tile, 96-col chunk).
// 8 waves = 2 M-groups (2 subtiles each) x 4 N-groups (3 tiles each).
// Each B fragment (2 x b128 global, K-contiguous after transpose) feeds
// 2 WMMAs. A fragments load straight from bf16 x copy (gathered rows).
// acc layout (ISA): row M = v + 8*(lane>>4), col N = lane&15.
// ---------------------------------------------------------------------------
__global__ __launch_bounds__(256) void moe_gateup(
    const u16* __restrict__ xbf, const u16* __restrict__ WgT,
    const u16* __restrict__ WuT, const int* __restrict__ cnt,
    const int* __restrict__ expOff, const int* __restrict__ tokmap,
    const float* __restrict__ wmap, u16* __restrict__ act) {

    const int e = blockIdx.x;
    const int count = cnt[e];
    const int base = blockIdx.y * 64;
    if (base >= count) return;

    const int tid = threadIdx.x, lane = tid & 31, wv = tid >> 5;
    const int kh = lane >> 4, ln = lane & 15;
    const int Mg = wv >> 2;                      // 0..1 -> M-subtiles {2Mg,2Mg+1}
    const int Ng = wv & 3;                       // 0..3 -> 48-col slice
    const int nb = blockIdx.z * 96 + Ng * 48;    // wave's global col base

    int tok[2];
#pragma unroll
    for (int s = 0; s < 2; ++s) {
        int slot = base + (Mg * 2 + s) * 16 + ln;
        tok[s] = (slot < count) ? tokmap[e * NTOK + slot] : 0;
    }

    v8f accG[3][2], accU[3][2];
#pragma unroll
    for (int j = 0; j < 3; ++j)
#pragma unroll
        for (int s = 0; s < 2; ++s) { v8f z = {}; accG[j][s] = z; accU[j][s] = z; }

    const u16* wg0 = WgT + ((size_t)e * INTER + nb + ln) * HID + kh * 16;
    const u16* wu0 = WuT + ((size_t)e * INTER + nb + ln) * HID + kh * 16;

    for (int k0 = 0; k0 < HID; k0 += 32) {
        FragAB a[2];
#pragma unroll
        for (int s = 0; s < 2; ++s) {
            const u16* ap = xbf + (size_t)tok[s] * HID + k0 + kh * 16;
            a[s].q[0] = *(const uint4*)ap;
            a[s].q[1] = *(const uint4*)(ap + 8);
        }
#pragma unroll
        for (int j = 0; j < 3; ++j) {
            const u16* gp = wg0 + (size_t)j * 16 * HID + k0;
            const u16* up = wu0 + (size_t)j * 16 * HID + k0;
            FragAB bg, bu;
            bg.q[0] = *(const uint4*)gp; bg.q[1] = *(const uint4*)(gp + 8);
            bu.q[0] = *(const uint4*)up; bu.q[1] = *(const uint4*)(up + 8);
            __builtin_prefetch(gp + 32, 0, 3);
            __builtin_prefetch(up + 32, 0, 3);
#pragma unroll
            for (int s = 0; s < 2; ++s) {
                accG[j][s] = __builtin_amdgcn_wmma_f32_16x16x32_bf16(
                    false, a[s].v, false, bg.v, (short)0, accG[j][s], false, false);
                accU[j][s] = __builtin_amdgcn_wmma_f32_16x16x32_bf16(
                    false, a[s].v, false, bu.v, (short)0, accU[j][s], false, false);
            }
        }
    }

    const int pairBase = expOff[e];
#pragma unroll
    for (int s = 0; s < 2; ++s) {
#pragma unroll
        for (int v = 0; v < 8; ++v) {
            int slot = base + (Mg * 2 + s) * 16 + v + kh * 8;
            if (slot < count) {
                float rw = wmap[e * NTOK + slot];
                size_t arow = (size_t)(pairBase + slot) * INTER;
#pragma unroll
                for (int j = 0; j < 3; ++j) {
                    float g = accG[j][s][v];
                    float u = accU[j][s][v];
                    float aa = (g / (1.f + __expf(-g))) * u * rw;
                    act[arow + nb + j * 16 + ln] = f2bf(aa);
                }
            }
        }
    }
}

// ---------------------------------------------------------------------------
// Down GEMM + scatter-add: block = (expert, 64-token tile, 256-col chunk).
// 8 waves = 2 M-groups x 4 N-groups (4 tiles each). A from bf16 act buffer.
// ---------------------------------------------------------------------------
__global__ __launch_bounds__(256) void moe_down(
    const u16* __restrict__ act, const u16* __restrict__ WdT,
    const int* __restrict__ cnt, const int* __restrict__ expOff,
    const int* __restrict__ tokmap, float* __restrict__ out) {

    const int e = blockIdx.x;
    const int count = cnt[e];
    const int base = blockIdx.y * 64;
    if (base >= count) return;

    const int tid = threadIdx.x, lane = tid & 31, wv = tid >> 5;
    const int kh = lane >> 4, ln = lane & 15;
    const int Mg = wv >> 2;
    const int Ng = wv & 3;
    const int nb = blockIdx.z * 256 + Ng * 64;
    const int pairBase = expOff[e];

    size_t arow[2];
#pragma unroll
    for (int s = 0; s < 2; ++s) {
        int slot = base + (Mg * 2 + s) * 16 + ln;
        if (slot >= count) slot = count - 1;     // clamp: rows discarded below
        arow[s] = (size_t)(pairBase + slot) * INTER;
    }

    v8f acc[4][2];
#pragma unroll
    for (int j = 0; j < 4; ++j)
#pragma unroll
        for (int s = 0; s < 2; ++s) { v8f z = {}; acc[j][s] = z; }

    const u16* wd0 = WdT + ((size_t)e * HID + nb + ln) * INTER + kh * 16;

    for (int k0 = 0; k0 < INTER; k0 += 32) {
        FragAB a[2];
#pragma unroll
        for (int s = 0; s < 2; ++s) {
            const u16* ap = act + arow[s] + k0 + kh * 16;
            a[s].q[0] = *(const uint4*)ap;
            a[s].q[1] = *(const uint4*)(ap + 8);
        }
#pragma unroll
        for (int j = 0; j < 4; ++j) {
            const u16* dp = wd0 + (size_t)j * 16 * INTER + k0;
            FragAB bd;
            bd.q[0] = *(const uint4*)dp; bd.q[1] = *(const uint4*)(dp + 8);
            __builtin_prefetch(dp + 32, 0, 3);
#pragma unroll
            for (int s = 0; s < 2; ++s) {
                acc[j][s] = __builtin_amdgcn_wmma_f32_16x16x32_bf16(
                    false, a[s].v, false, bd.v, (short)0, acc[j][s], false, false);
            }
        }
    }

#pragma unroll
    for (int s = 0; s < 2; ++s) {
#pragma unroll
        for (int v = 0; v < 8; ++v) {
            int slot = base + (Mg * 2 + s) * 16 + v + kh * 8;
            if (slot < count) {
                int t = tokmap[e * NTOK + slot];
                float* orow = out + (size_t)t * HID + nb;
#pragma unroll
                for (int j = 0; j < 4; ++j)
                    atomicAdd(&orow[j * 16 + ln], acc[j][s][v]);
            }
        }
    }
}

// ---------------------------------------------------------------------------
extern "C" void kernel_launch(void* const* d_in, const int* in_sizes, int n_in,
                              void* d_out, int out_size, void* d_ws, size_t ws_size,
                              hipStream_t stream) {
    const float* x  = (const float*)d_in[0];   // [1,2048,1024]
    const float* gw = (const float*)d_in[1];   // [32,1024]
    const float* Wg = (const float*)d_in[2];   // [32,1024,768]
    const float* Wu = (const float*)d_in[3];   // [32,1024,768]
    const float* Wd = (const float*)d_in[4];   // [32,768,1024]
    float* out = (float*)d_out;                // [1,2048,1024]

    char* ws = (char*)d_ws;
    // workspace layout (bytes)
    const size_t OFF_CNT  = 0;                               // 32 ints
    const size_t OFF_EOFF = 128;                             // 32 ints
    const size_t OFF_TOK  = 256;                             // 32*2048 ints
    const size_t OFF_WMAP = OFF_TOK  + sizeof(int)   * NEXP * NTOK;
    const size_t OFF_XBF  = OFF_WMAP + sizeof(float) * NEXP * NTOK;
    const size_t OFF_ACT  = OFF_XBF  + sizeof(u16) * (size_t)NTOK * HID;
    const size_t OFF_WGT  = OFF_ACT  + sizeof(u16) * (size_t)(NTOK * TOPK + 64) * INTER;
    const size_t OFF_WUT  = OFF_WGT  + sizeof(u16) * (size_t)NEXP * HID * INTER;
    const size_t OFF_WDT  = OFF_WUT  + sizeof(u16) * (size_t)NEXP * HID * INTER;

    int*   cnt    = (int*)(ws + OFF_CNT);
    int*   expOff = (int*)(ws + OFF_EOFF);
    int*   tokmap = (int*)(ws + OFF_TOK);
    float* wmap   = (float*)(ws + OFF_WMAP);
    u16*   xbf    = (u16*)(ws + OFF_XBF);
    u16*   actb   = (u16*)(ws + OFF_ACT);
    u16*   WgT    = (u16*)(ws + OFF_WGT);
    u16*   WuT    = (u16*)(ws + OFF_WUT);
    u16*   WdT    = (u16*)(ws + OFF_WDT);

    moe_zero<<<(NTOK * HID + 255) / 256, 256, 0, stream>>>(out, cnt);
    moe_cvt_x<<<(NTOK * HID) / (256 * 8), 256, 0, stream>>>(x, xbf);

    dim3 tgu(INTER / 64, HID / 64, NEXP);      // Wg/Wu: R=HID rows, C=INTER cols
    moe_transpose_cvt<<<tgu, 256, 0, stream>>>(Wg, WgT, HID, INTER);
    moe_transpose_cvt<<<tgu, 256, 0, stream>>>(Wu, WuT, HID, INTER);
    dim3 td(HID / 64, INTER / 64, NEXP);       // Wd: R=INTER rows, C=HID cols
    moe_transpose_cvt<<<td, 256, 0, stream>>>(Wd, WdT, INTER, HID);

    moe_router<<<NTOK / 8, 256, 0, stream>>>(x, gw, cnt, tokmap, wmap);
    moe_offsets<<<1, 32, 0, stream>>>(cnt, expOff);

    dim3 ggu(NEXP, NTOK / 64, INTER / 96);     // (32, 32, 8)
    moe_gateup<<<ggu, 256, 0, stream>>>(xbf, WgT, WuT, cnt, expOff, tokmap, wmap, actb);

    dim3 gdn(NEXP, NTOK / 64, HID / 256);      // (32, 32, 4)
    moe_down<<<gdn, 256, 0, stream>>>(actb, WdT, cnt, expOff, tokmap, out);
}
